// GraphClassifier_65755949301918
// MI455X (gfx1250) — matrix operand order repaired and checked by reading, not compiled, not run on hardware
//
#include <hip/hip_runtime.h>
#include <hip/hip_bf16.h>

// ---------------------------------------------------------------------------
// GraphClassifier for MI455X (gfx1250, wave32, WMMA 16x16x32 f16)
//
// B=4, N=512, NF=12, EF=1, QK=128, NV_D=128, EV_D=32, TR=256, NCLS=5
//
// Key transforms:
//  * EV (4,512,512,32) is never materialized: edge_part = Wev*S1 + bev*S0
//    with S1 = sum_j w*edges, S0 = sum_j w  (rank-1 collapse, saves ~134MB).
//  * scores = Q K^T, node_part = w NV, Trans GEMM all run on
//    v_wmma_f32_16x16x32_f16 (f32 accumulate).
//  * NV stored transposed (d-major) so WMMA B-fragments are contiguous.
//  * BatchNorm batch statistics via global f32 atomics + follow-up kernel.
//  * All divisions/tanh/rsqrt use raw TRANS ops (v_rcp/v_exp/v_rsq), no IEEE
//    divide expansion: these co-execute with the WMMA pipeline.
// ---------------------------------------------------------------------------

typedef _Float16 v16h __attribute__((ext_vector_type(16)));
typedef _Float16 v8h  __attribute__((ext_vector_type(8)));
typedef float    v8f  __attribute__((ext_vector_type(8)));

#define BB   4
#define NN   512
#define NF   12
#define QKD  128
#define NVD  128
#define EVD  32
#define TRD  256
#define KTR  160            // NVD + EVD
#define NROWS (BB*NN)       // 2048

// ---- workspace layout (bytes), all 256-aligned -----------------------------
#define OFF_QH   0u          // 2048*128 f16 = 512KB
#define OFF_KH   524288u     // 512KB
#define OFF_NVT  1048576u    // 4*128*512 f16 = 512KB (transposed NV)
#define OFF_WTH  1572864u    // 256*160 f16 = 80KB
#define OFF_VALS 1654784u    // 2048*256 f32 = 2MB
#define OFF_SUM  3751936u    // 256 f32 sum + 256 f32 sumsq
// total ~3.58MB

// ---- fast TRANS-op math ----------------------------------------------------
__device__ inline float fast_rcp(float x) { return __builtin_amdgcn_rcpf(x); }
__device__ inline float fast_tanh(float x) {
  // tanh(x) = 1 - 2/(exp(2x)+1); clamp to keep exp finite
  float xc = fminf(fmaxf(x, -10.0f), 10.0f);
  float e2 = __expf(2.0f * xc);
  return 1.0f - 2.0f * fast_rcp(e2 + 1.0f);
}

// ---- wave reductions (wave32) ---------------------------------------------
__device__ inline float wred_sum(float x) {
  #pragma unroll
  for (int o = 16; o > 0; o >>= 1) x += __shfl_xor(x, o, 32);
  return x;
}
__device__ inline float wred_max(float x) {
  #pragma unroll
  for (int o = 16; o > 0; o >>= 1) x = fmaxf(x, __shfl_xor(x, o, 32));
  return x;
}

// ---- WMMA fragment loaders (16-bit, 16x16x32) ------------------------------
// A (16x32): lane<16 holds row=lane, K = kk+{0..7} and kk+{16..23};
//            lane>=16 holds row=lane-16, K = kk+{8..15} and kk+{24..31}.
template <typename P>
__device__ inline v16h ldA(P rowptr, int kk, int hi) {
  v8h lo = *(const v8h*)(rowptr + kk + hi * 8);
  v8h hc = *(const v8h*)(rowptr + kk + 16 + hi * 8);
  return __builtin_shufflevector(lo, hc, 0,1,2,3,4,5,6,7,8,9,10,11,12,13,14,15);
}
// B (32x16): lane<16 holds col=lane, K = kk+{0..7} and kk+{16..23};
//            lane>=16 col=lane-16, K = kk+{8..15} and kk+{24..31}.
template <typename P>
__device__ inline v16h ldB(P rowptr, int kk, int hi) {
  v8h lo = *(const v8h*)(rowptr + kk + hi * 8);
  v8h hc = *(const v8h*)(rowptr + kk + 16 + hi * 8);
  return __builtin_shufflevector(lo, hc, 0,1,2,3,4,5,6,7,8,9,10,11,12,13,14,15);
}

__device__ inline v8f wmma_f16(v16h a, v16h b, v8f c) {
  return __builtin_amdgcn_wmma_f32_16x16x32_f16(false, a, false, b,
                                                (short)0, c, false, false);
}

// ===========================================================================
// Kernel 0: convert Wt -> f16, zero batchnorm accumulators
// ===========================================================================
__global__ __launch_bounds__(256) void gc_init_kernel(
    const float* __restrict__ Wt, _Float16* __restrict__ Wth,
    float* __restrict__ sums /* 512 floats: sum[256] ++ sumsq[256] */) {
  int g = blockIdx.x * 256 + threadIdx.x;
  if (g < TRD * KTR) Wth[g] = (_Float16)Wt[g];
  if (g < 512) sums[g] = 0.0f;
}

// ===========================================================================
// Kernel 1: projections Q,K (row-major f16) and NV (transposed f16)
// one thread per (row, d) pair; K=12 dot products
// ===========================================================================
__global__ __launch_bounds__(256) void gc_proj_kernel(
    const float* __restrict__ nodes,
    const float* __restrict__ Wq, const float* __restrict__ bq,
    const float* __restrict__ Wk, const float* __restrict__ bk,
    const float* __restrict__ Wnv, const float* __restrict__ bnv,
    _Float16* __restrict__ Qh, _Float16* __restrict__ Kh,
    _Float16* __restrict__ NVt) {
  int g = blockIdx.x * 256 + threadIdx.x;      // 0 .. 2048*128-1
  int row = g >> 7;
  int d   = g & 127;
  int b   = row >> 9;
  int i   = row & 511;
  const float* nr = nodes + (size_t)row * NF;
  const float* wq = Wq  + d * NF;
  const float* wk = Wk  + d * NF;
  const float* wv = Wnv + d * NF;
  float q = 0.f, k = 0.f, v = 0.f;
  #pragma unroll
  for (int t = 0; t < NF; ++t) {
    float x = nr[t];
    q += x * wq[t]; k += x * wk[t]; v += x * wv[t];
  }
  Qh[(size_t)row * QKD + d] = (_Float16)(q + bq[d]);
  Kh[(size_t)row * QKD + d] = (_Float16)(k + bk[d]);
  NVt[((size_t)(b * NVD + d)) * NN + i] = (_Float16)(v + bnv[d]);
}

// ===========================================================================
// Kernel 2: fused attention for one (batch, 16-row tile) per wave32
//   scores (WMMA) -> softmax (LDS) -> node_part (WMMA) + edge rank-1
//   -> tanh -> Trans GEMM (WMMA) -> vals + BN atomics
// ===========================================================================
__global__ __launch_bounds__(32) void gc_attn_kernel(
    const float* __restrict__ edges, const float* __restrict__ distances,
    const int* __restrict__ mask,
    const _Float16* __restrict__ Qh, const _Float16* __restrict__ Kh,
    const _Float16* __restrict__ NVt, const _Float16* __restrict__ Wth,
    const float* __restrict__ Wev, const float* __restrict__ bev,
    const float* __restrict__ bt,
    float* __restrict__ vals, float* __restrict__ sum,
    float* __restrict__ sumsq) {
  const int lane = threadIdx.x & 31;
  const int lr = lane & 15;
  const int hi = lane >> 4;
  const int tile = blockIdx.x;          // 0..127
  const int b  = tile >> 5;             // 32 tiles per batch
  const int i0 = (tile & 31) * 16;

  __shared__ float    sS[16 * NN];      // 32KB scores; f16 w overlays front
  __shared__ _Float16 sT[16 * KTR];     // tanh activations 16x160
  __shared__ float    sS0[16], sS1[16]; // per-row edge sums
  _Float16* sW = (_Float16*)sS;         // w rows: stride 512 halfs (1KB)

  const float coeff = 0.08838834764831845f;   // 1/sqrt(128)

  // ---- Q fragments for this row tile (reused across all 32 j-tiles) ----
  const _Float16* qrow = Qh + (size_t)(b * NN + i0 + lr) * QKD;
  v16h aq[4];
  #pragma unroll
  for (int k = 0; k < 4; ++k) aq[k] = ldA(qrow, k * 32, hi);

  // ---- scores = Q K^T, weighted + masked, into LDS ----
  for (int jt = 0; jt < 32; ++jt) {
    const int j0 = jt * 16;
    const _Float16* krow = Kh + (size_t)(b * NN + j0 + lr) * QKD;
    v8f c = {};
    #pragma unroll
    for (int k = 0; k < 4; ++k) c = wmma_f16(aq[k], ldB(krow, k * 32, hi), c);

    const int jcol = j0 + lr;
    if (jt + 1 < 32)  // warm L2 for next distance columns
      __builtin_prefetch(distances + ((size_t)(b * NN + i0) * NN) + jcol + 16, 0, 1);
    #pragma unroll
    for (int r = 0; r < 8; ++r) {
      const int row = r + hi * 8;
      const size_t idx = ((size_t)(b * NN + i0 + row)) * NN + jcol;
      float s = c[r] * coeff * fast_rcp(distances[idx] + 1.0f);
      if (mask[idx] != 0) s = -__builtin_inff();
      sS[row * NN + jcol] = s;
    }
  }

  // ---- softmax rows + edge sums; write w as f16 (LDS overlay reuse) ----
  for (int r = 0; r < 16; ++r) {
    float v[16];
    #pragma unroll
    for (int t = 0; t < 16; ++t) v[t] = sS[r * NN + t * 32 + lane];
    float m = -__builtin_inff();
    #pragma unroll
    for (int t = 0; t < 16; ++t) m = fmaxf(m, v[t]);
    m = wred_max(m);
    float e[16], ps = 0.f;
    #pragma unroll
    for (int t = 0; t < 16; ++t) {
      e[t] = (v[t] < -1e37f) ? 0.0f : __expf(v[t] - m);
      ps += e[t];
    }
    const float s = wred_sum(ps);
    const float inv = (s > 0.0f) ? fast_rcp(s) : 0.0f;  // fully-masked row -> 0
    const float* erow = edges + ((size_t)(b * NN + i0 + r)) * NN;
    float p1 = 0.f;
    #pragma unroll
    for (int t = 0; t < 16; ++t) {
      const float w = e[t] * inv;
      sW[r * NN + t * 32 + lane] = (_Float16)w;
      p1 += w * erow[t * 32 + lane];
    }
    const float s1 = wred_sum(p1);
    if (lane == 0) { sS0[r] = (s > 0.0f) ? 1.0f : 0.0f; sS1[r] = s1; }
  }

  // ---- node_part = w @ NV  (K=512), tanh into sT[:, 0:128] ----
  for (int nt = 0; nt < 8; ++nt) {
    const int n0 = nt * 16;
    const _Float16* nvrow = NVt + (size_t)(b * NVD + n0 + lr) * NN;
    v8f c = {};
    for (int kk = 0; kk < NN; kk += 32)
      c = wmma_f16(ldA(sW + lr * NN, kk, hi), ldB(nvrow, kk, hi), c);
    #pragma unroll
    for (int r = 0; r < 8; ++r)
      sT[(r + hi * 8) * KTR + n0 + lr] = (_Float16)fast_tanh(c[r]);
  }

  // ---- edge_part rank-1: d = lane (0..31), tanh into sT[:, 128:160] ----
  {
    const float we = Wev[lane];
    const float be = bev[lane];
    for (int r = 0; r < 16; ++r)
      sT[r * KTR + NVD + lane] = (_Float16)fast_tanh(we * sS1[r] + be * sS0[r]);
  }

  // ---- Trans GEMM: (16x160) @ Wt^T -> (16x256), + BN accumulation ----
  for (int ct = 0; ct < 16; ++ct) {
    const int c0 = ct * 16;
    const _Float16* wtrow = Wth + (size_t)(c0 + lr) * KTR;
    v8f c = {};
    #pragma unroll
    for (int kk = 0; kk < KTR; kk += 32)
      c = wmma_f16(ldA(sT + lr * KTR, kk, hi), ldB(wtrow, kk, hi), c);

    const int ch = c0 + lr;
    const float btc = bt[ch];
    float psum = 0.f, psq = 0.f;
    #pragma unroll
    for (int r = 0; r < 8; ++r) {
      const int row = r + hi * 8;
      const float v = c[r] + btc;
      vals[((size_t)(b * NN + i0 + row)) * TRD + ch] = v;
      psum += v; psq += v * v;
    }
    psum += __shfl_xor(psum, 16, 32);   // fold the two half-wave row groups
    psq  += __shfl_xor(psq, 16, 32);
    if (hi == 0) {
      atomicAdd(&sum[ch], psum);
      atomicAdd(&sumsq[ch], psq);
    }
  }
}

// ===========================================================================
// Kernel 3: batchnorm (batch stats) + classifier head; one wave per row
// ===========================================================================
__global__ __launch_bounds__(256) void gc_bncls_kernel(
    const float* __restrict__ vals, const float* __restrict__ sum,
    const float* __restrict__ sumsq, const float* __restrict__ gamma,
    const float* __restrict__ beta, const float* __restrict__ Wc,
    const float* __restrict__ bc, float* __restrict__ out) {
  const int lane = threadIdx.x & 31;
  const int row = blockIdx.x * 8 + (threadIdx.x >> 5);
  const float invN = 1.0f / (float)NROWS;
  float acc[5] = {0.f, 0.f, 0.f, 0.f, 0.f};
  #pragma unroll
  for (int t = 0; t < 8; ++t) {
    const int ch = t * 32 + lane;
    const float mu = sum[ch] * invN;
    const float var = sumsq[ch] * invN - mu * mu;
    const float rs = __builtin_amdgcn_rsqf(var + 1e-5f);
    const float nv =
        (vals[(size_t)row * TRD + ch] - mu) * rs * gamma[ch] + beta[ch];
    #pragma unroll
    for (int c = 0; c < 5; ++c) acc[c] += nv * Wc[c * TRD + ch];
  }
  #pragma unroll
  for (int c = 0; c < 5; ++c) acc[c] = wred_sum(acc[c]);
  if (lane == 0) {
    #pragma unroll
    for (int c = 0; c < 5; ++c) out[(size_t)row * 5 + c] = acc[c] + bc[c];
  }
}

// ===========================================================================
extern "C" void kernel_launch(void* const* d_in, const int* in_sizes, int n_in,
                              void* d_out, int out_size, void* d_ws,
                              size_t ws_size, hipStream_t stream) {
  const float* nodes = (const float*)d_in[0];
  const float* edges = (const float*)d_in[1];
  const float* dist  = (const float*)d_in[2];
  const int*   mask  = (const int*)d_in[3];
  const float* Wq  = (const float*)d_in[4];
  const float* bq  = (const float*)d_in[5];
  const float* Wk  = (const float*)d_in[6];
  const float* bk  = (const float*)d_in[7];
  const float* Wnv = (const float*)d_in[8];
  const float* bnv = (const float*)d_in[9];
  const float* Wev = (const float*)d_in[10];
  const float* bev = (const float*)d_in[11];
  const float* Wt  = (const float*)d_in[12];
  const float* bt  = (const float*)d_in[13];
  const float* gamma = (const float*)d_in[14];
  const float* beta  = (const float*)d_in[15];
  const float* Wc  = (const float*)d_in[16];
  const float* bc  = (const float*)d_in[17];
  float* out = (float*)d_out;

  char* ws = (char*)d_ws;
  _Float16* Qh   = (_Float16*)(ws + OFF_QH);
  _Float16* Kh   = (_Float16*)(ws + OFF_KH);
  _Float16* NVt  = (_Float16*)(ws + OFF_NVT);
  _Float16* Wth  = (_Float16*)(ws + OFF_WTH);
  float*    vals = (float*)(ws + OFF_VALS);
  float*    sums = (float*)(ws + OFF_SUM);    // [0,256) sum, [256,512) sumsq

  gc_init_kernel<<<(TRD * KTR + 255) / 256, 256, 0, stream>>>(Wt, Wth, sums);
  gc_proj_kernel<<<(NROWS * QKD) / 256, 256, 0, stream>>>(
      nodes, Wq, bq, Wk, bk, Wnv, bnv, Qh, Kh, NVt);
  gc_attn_kernel<<<BB * (NN / 16), 32, 0, stream>>>(
      edges, dist, mask, Qh, Kh, NVt, Wth, Wev, bev, bt, vals, sums,
      sums + 256);
  gc_bncls_kernel<<<NROWS / 8, 256, 0, stream>>>(
      vals, sums, sums + 256, gamma, beta, Wc, bc, out);
}
